// DIEN_49744311222900
// MI455X (gfx1250) — matrix operand order repaired and checked by reading, not compiled, not run on hardware
//
#include <hip/hip_runtime.h>
#include <hip/hip_bf16.h>
#include <stdint.h>
#include <stddef.h>

typedef __attribute__((ext_vector_type(16))) __bf16       v16bf;
typedef __attribute__((ext_vector_type(8)))  float        v8f;
typedef __attribute__((ext_vector_type(4)))  unsigned int v4u;
typedef __attribute__((ext_vector_type(8)))  int          v8i_;
typedef __attribute__((ext_vector_type(4)))  int          v4i_;

#define B_SZ 2048
#define T_SZ 128
#define D_SZ 64
#define H_SZ 64
#define ROWS 16    // batch rows per block (one WMMA M tile)
#define LDH  72    // bf16 LDS row stride for h / r*h tiles (bank-conflict pad)
#define LDXF 66    // f32 row stride of TDM-padded behaviors tile (64 + 2 pad DW)
#define LDSS 68    // bf16 row stride of TDM-padded seq tile (64 + 4 pad bf16)
#define NTHR 128   // 4 waves; each wave owns a 16-wide column slice of H

// TDM D# group1 config dword0:
//   data_size<<16 | pad_enable<<20 | pad_interval<<22 | pad_amount<<25
#define TDM_CFG_F32  (int)((2u << 16) | (1u << 20) | (5u << 22) | (1u << 25)) // pad 2DW / 64DW row
#define TDM_CFG_BF16 (int)((1u << 16) | (1u << 20) | (4u << 22) | (1u << 25)) // pad 2DW / 32DW row

__device__ __forceinline__ v8f wmma_bf16(v16bf a, v16bf b, v8f c) {
  return __builtin_amdgcn_wmma_f32_16x16x32_bf16(false, a, false, b, (short)0, c,
                                                 false, false);
}

__device__ __forceinline__ float sigmoidf_(float x) {
  return 1.0f / (1.0f + __expf(-x));
}

// Issue one TDM tile load: 16 rows x 64 elements, global row stride 8192
// elements, into LDS at lds_off with per-row padding (see TDM_CFG_*).
__device__ __forceinline__ void tdm_load_tile(unsigned lds_off, const void* gptr,
                                              int cfg0) {
  unsigned long long ga = (unsigned long long)(uintptr_t)gptr;
  v4u g0;
  g0[0] = 1u;                                   // count=1, user descriptor
  g0[1] = lds_off;                              // lds_addr (bytes)
  g0[2] = (unsigned)(ga & 0xffffffffu);         // global_addr[31:0]
  g0[3] = (unsigned)(ga >> 32) | 0x80000000u;   // global_addr[56:32] | type=2
  v8i_ g1;
  g1[0] = cfg0;                                 // data_size / pad config
  g1[1] = (int)(64u << 16);                     // tensor_dim0 = 64
  g1[2] = 0;                                    // tensor_dim1[15:0] = 0 (of 1<<16)
  g1[3] = (int)((64u << 16) | 1u);              // tile_dim0=64 | tensor_dim1 hi
  g1[4] = 16;                                   // tile_dim1 = 16 rows
  g1[5] = 8192;                                 // tensor_dim0_stride = T*64
  g1[6] = 0;
  g1[7] = 0;
  v4i_ z4 = {0, 0, 0, 0};                       // 2D tensor: groups 2/3 unused
  v8i_ z8 = {0, 0, 0, 0, 0, 0, 0, 0};
  __builtin_amdgcn_tensor_load_to_lds(g0, g1, z4, z4, z8, 0);
}

// B-matrix fragment (KxN, row-major f32 in global, ld = row stride).
__device__ __forceinline__ v16bf load_b_frag(const float* W, int ld, int col,
                                             int k0, int lane) {
  int kb = k0 + ((lane & 16) ? 8 : 0);
  const float* p = W + (size_t)kb * ld + col;
  v16bf b;
#pragma unroll
  for (int j = 0; j < 8; ++j) { b[j] = (__bf16)p[(size_t)j * ld]; b[8 + j] = (__bf16)p[(size_t)(16 + j) * ld]; }
  return b;
}

// A-matrix fragment from LDS bf16, row-major [16][stride].
__device__ __forceinline__ v16bf load_a_lds(const __bf16* A, int stride, int row,
                                            int k0, int lane) {
  int kb = k0 + ((lane & 16) ? 8 : 0);
  const __bf16* p = A + row * stride + kb;
  v16bf a;
#pragma unroll
  for (int j = 0; j < 8; ++j) { a[j] = p[j]; a[8 + j] = p[16 + j]; }
  return a;
}

// A-matrix fragment from LDS f32 (TDM-staged behaviors tile), cvt to bf16.
__device__ __forceinline__ v16bf load_a_lds_f32(const float* A, int stride,
                                                int row, int k0, int lane) {
  int kb = k0 + ((lane & 16) ? 8 : 0);
  const float* p = A + row * stride + kb;
  v16bf a;
#pragma unroll
  for (int j = 0; j < 8; ++j) { a[j] = (__bf16)p[j]; a[8 + j] = (__bf16)p[16 + j]; }
  return a;
}

// A-matrix fragment straight from global f32 (candidate, used once).
__device__ __forceinline__ v16bf load_a_glb(const float* A, int stride, int row,
                                            int k0, int lane) {
  int kb = k0 + ((lane & 16) ? 8 : 0);
  const float* p = A + (size_t)row * stride + kb;
  v16bf a;
#pragma unroll
  for (int j = 0; j < 8; ++j) { a[j] = (__bf16)p[j]; a[8 + j] = (__bf16)p[16 + j]; }
  return a;
}

__global__ __launch_bounds__(NTHR) void dien_fused(
    const float* __restrict__ behaviors, const float* __restrict__ candidate,
    const unsigned char* __restrict__ mask,
    const float* __restrict__ gru_kernel, const float* __restrict__ gru_rec,
    const float* __restrict__ gru_bias,
    const float* __restrict__ W_kernel, const float* __restrict__ W_bias,
    const float* __restrict__ Wxu, const float* __restrict__ bxu,
    const float* __restrict__ Whu,
    const float* __restrict__ Wxr, const float* __restrict__ bxr,
    const float* __restrict__ Whr,
    const float* __restrict__ Wxg, const float* __restrict__ bxg,
    const float* __restrict__ Whg,
    float* __restrict__ out, void* __restrict__ ws) {
  __shared__ __align__(16) float  sh_xf[2][ROWS * LDXF];  // TDM x_t double buffer
  __shared__ __align__(16) __bf16 sh_s[2][ROWS * LDSS];   // TDM s_t double buffer
  __shared__ __bf16 sh_h[ROWS * LDH];    // hidden state (bf16, WMMA A source)
  __shared__ __bf16 sh_rh[ROWS * LDH];   // r*h tile (bf16)
  __shared__ float  sh_hf[ROWS * H_SZ];  // hidden state (f32, elementwise)
  __shared__ float  sh_aw[ROWS * H_SZ];  // attention query per row
  __shared__ float  sh_sc[T_SZ * ROWS];  // scores -> normalized attention

  __bf16* __restrict__ seq_ws = (__bf16*)ws;  // [B, T, H] bf16 GRU outputs

  const int tid  = threadIdx.x;
  const int lane = tid & 31;
  const int wave = tid >> 5;
  const int cl   = wave * 16 + (lane & 15);
  const int mrow = (lane & 16) ? 8 : 0;
  const int arow = lane & 15;
  const int b0   = blockIdx.x * ROWS;

  // ---- Phase 1: aw = candidate @ W_kernel + W_bias ------------------------
  {
    v16bf wk0 = load_b_frag(W_kernel, H_SZ, cl, 0, lane);
    v16bf wk1 = load_b_frag(W_kernel, H_SZ, cl, 32, lane);
    v16bf ca0 = load_a_glb(candidate, D_SZ, b0 + arow, 0, lane);
    v16bf ca1 = load_a_glb(candidate, D_SZ, b0 + arow, 32, lane);
    v8f acc = {};
    acc = wmma_bf16(ca0, wk0, acc);
    acc = wmma_bf16(ca1, wk1, acc);
    float wb = W_bias[cl];
#pragma unroll
    for (int r = 0; r < 8; ++r) sh_aw[(mrow + r) * H_SZ + cl] = acc[r] + wb;
  }
  for (int i = tid; i < ROWS * LDH; i += NTHR) sh_h[i] = (__bf16)0.0f;
  for (int i = tid; i < ROWS * H_SZ; i += NTHR) sh_hf[i] = 0.0f;
  __syncthreads();

  // Prefetch tile t=0 via Tensor Data Mover while weight fragments load.
  if (wave == 0)
    tdm_load_tile((unsigned)(uintptr_t)&sh_xf[0][0],
                  behaviors + (size_t)b0 * T_SZ * D_SZ, TDM_CFG_F32);

  // ---- GRU weight fragments: persist in VGPRs across the whole scan -------
  const v16bf gz0 = load_b_frag(gru_kernel, 3 * H_SZ, cl,            0,  lane);
  const v16bf gz1 = load_b_frag(gru_kernel, 3 * H_SZ, cl,            32, lane);
  const v16bf gr0 = load_b_frag(gru_kernel, 3 * H_SZ, H_SZ + cl,     0,  lane);
  const v16bf gr1 = load_b_frag(gru_kernel, 3 * H_SZ, H_SZ + cl,     32, lane);
  const v16bf gh0 = load_b_frag(gru_kernel, 3 * H_SZ, 2 * H_SZ + cl, 0,  lane);
  const v16bf gh1 = load_b_frag(gru_kernel, 3 * H_SZ, 2 * H_SZ + cl, 32, lane);
  const v16bf uz0 = load_b_frag(gru_rec,    3 * H_SZ, cl,            0,  lane);
  const v16bf uz1 = load_b_frag(gru_rec,    3 * H_SZ, cl,            32, lane);
  const v16bf ur0 = load_b_frag(gru_rec,    3 * H_SZ, H_SZ + cl,     0,  lane);
  const v16bf ur1 = load_b_frag(gru_rec,    3 * H_SZ, H_SZ + cl,     32, lane);
  const v16bf uh0 = load_b_frag(gru_rec,    3 * H_SZ, 2 * H_SZ + cl, 0,  lane);
  const v16bf uh1 = load_b_frag(gru_rec,    3 * H_SZ, 2 * H_SZ + cl, 32, lane);
  const float bz = gru_bias[cl];
  const float br = gru_bias[H_SZ + cl];
  const float bh = gru_bias[2 * H_SZ + cl];

  // ---- Phase 2: GRU scan (x-projection fused; TDM double-buffered) --------
  for (int t = 0; t < T_SZ; ++t) {
    if (wave == 0) {
      if (t + 1 < T_SZ) {  // stream next tile while this step computes
        tdm_load_tile((unsigned)(uintptr_t)&sh_xf[(t + 1) & 1][0],
                      behaviors + ((size_t)b0 * T_SZ + (t + 1)) * D_SZ,
                      TDM_CFG_F32);
        __builtin_amdgcn_s_wait_tensorcnt(1);  // tile t resident
      } else {
        __builtin_amdgcn_s_wait_tensorcnt(0);
      }
    }
    __syncthreads();  // tile t + h(t-1) visible to all waves
    const float* xt = sh_xf[t & 1];

    v16bf xa0 = load_a_lds_f32(xt, LDXF, arow, 0, lane);
    v16bf xa1 = load_a_lds_f32(xt, LDXF, arow, 32, lane);
    v16bf ha0 = load_a_lds(sh_h, LDH, arow, 0, lane);
    v16bf ha1 = load_a_lds(sh_h, LDH, arow, 32, lane);

    v8f az, ar;
#pragma unroll
    for (int r = 0; r < 8; ++r) { az[r] = bz; ar[r] = br; }
    az = wmma_bf16(xa0, gz0, az); az = wmma_bf16(xa1, gz1, az);
    az = wmma_bf16(ha0, uz0, az); az = wmma_bf16(ha1, uz1, az);
    ar = wmma_bf16(xa0, gr0, ar); ar = wmma_bf16(xa1, gr1, ar);
    ar = wmma_bf16(ha0, ur0, ar); ar = wmma_bf16(ha1, ur1, ar);

    float zv[8], hold[8];
#pragma unroll
    for (int r = 0; r < 8; ++r) {
      int M = mrow + r;
      hold[r] = sh_hf[M * H_SZ + cl];
      zv[r] = sigmoidf_(az[r]);
      float rv = sigmoidf_(ar[r]);
      sh_rh[M * LDH + cl] = (__bf16)(rv * hold[r]);
    }
    __syncthreads();  // all h reads done; rh tile visible

    v16bf ra0 = load_a_lds(sh_rh, LDH, arow, 0, lane);
    v16bf ra1 = load_a_lds(sh_rh, LDH, arow, 32, lane);
    v8f ah;
#pragma unroll
    for (int r = 0; r < 8; ++r) ah[r] = bh;
    ah = wmma_bf16(xa0, gh0, ah); ah = wmma_bf16(xa1, gh1, ah);
    ah = wmma_bf16(ra0, uh0, ah); ah = wmma_bf16(ra1, uh1, ah);

#pragma unroll
    for (int r = 0; r < 8; ++r) {
      int M = mrow + r;
      float hh = tanhf(ah[r]);
      float hn = zv[r] * hold[r] + (1.0f - zv[r]) * hh;
      unsigned char m = mask[(size_t)(b0 + M) * T_SZ + t];
      hn = m ? hn : hold[r];
      sh_hf[M * H_SZ + cl] = hn;
      sh_h[M * LDH + cl] = (__bf16)hn;
      seq_ws[((size_t)(b0 + M) * T_SZ + t) * H_SZ + cl] = (__bf16)hn;
    }
    __syncthreads();

    // incremental attention score: exp(dot(aw[b], h_t[b])) (masked)
    if (tid < ROWS) {
      float s = 0.0f;
#pragma unroll
      for (int c = 0; c < H_SZ; ++c)
        s += sh_aw[tid * H_SZ + c] * sh_hf[tid * H_SZ + c];
      unsigned char m = mask[(size_t)(b0 + tid) * T_SZ + t];
      sh_sc[t * ROWS + tid] = m ? __expf(s) : 0.0f;
    }
  }
  // seq stores must be visible in L2 before the TDM reads them back
  asm volatile("s_wait_storecnt 0x0" ::: "memory");
  __syncthreads();

  // ---- Phase 3: normalize attention, reset hidden state -------------------
  if (tid < ROWS) {
    float den = 1e-8f;
    for (int t = 0; t < T_SZ; ++t) den += sh_sc[t * ROWS + tid];
    float inv = 1.0f / den;
    for (int t = 0; t < T_SZ; ++t) sh_sc[t * ROWS + tid] *= inv;
  }
  for (int i = tid; i < ROWS * LDH; i += NTHR) sh_h[i] = (__bf16)0.0f;
  for (int i = tid; i < ROWS * H_SZ; i += NTHR) sh_hf[i] = 0.0f;
  __syncthreads();

  if (wave == 0)
    tdm_load_tile((unsigned)(uintptr_t)&sh_s[0][0],
                  seq_ws + (size_t)b0 * T_SZ * H_SZ, TDM_CFG_BF16);

  // ---- AUGRU weight fragments (GRU fragments dead -> regs reused) ---------
  const v16bf xu0 = load_b_frag(Wxu, H_SZ, cl, 0, lane);
  const v16bf xu1 = load_b_frag(Wxu, H_SZ, cl, 32, lane);
  const v16bf hu0 = load_b_frag(Whu, H_SZ, cl, 0, lane);
  const v16bf hu1 = load_b_frag(Whu, H_SZ, cl, 32, lane);
  const v16bf xr0 = load_b_frag(Wxr, H_SZ, cl, 0, lane);
  const v16bf xr1 = load_b_frag(Wxr, H_SZ, cl, 32, lane);
  const v16bf hr0 = load_b_frag(Whr, H_SZ, cl, 0, lane);
  const v16bf hr1 = load_b_frag(Whr, H_SZ, cl, 32, lane);
  const v16bf xg0 = load_b_frag(Wxg, H_SZ, cl, 0, lane);
  const v16bf xg1 = load_b_frag(Wxg, H_SZ, cl, 32, lane);
  const v16bf hg0 = load_b_frag(Whg, H_SZ, cl, 0, lane);
  const v16bf hg1 = load_b_frag(Whg, H_SZ, cl, 32, lane);
  const float bu = bxu[cl];
  const float brr = bxr[cl];
  const float bg = bxg[cl];

  // ---- Phase 4: AUGRU scan (TDM double-buffered seq tiles) ----------------
  for (int t = 0; t < T_SZ; ++t) {
    if (wave == 0) {
      if (t + 1 < T_SZ) {
        tdm_load_tile((unsigned)(uintptr_t)&sh_s[(t + 1) & 1][0],
                      seq_ws + ((size_t)b0 * T_SZ + (t + 1)) * H_SZ,
                      TDM_CFG_BF16);
        __builtin_amdgcn_s_wait_tensorcnt(1);
      } else {
        __builtin_amdgcn_s_wait_tensorcnt(0);
      }
    }
    __syncthreads();  // tile t + h(t-1) visible
    const __bf16* st = sh_s[t & 1];

    v16bf sa0 = load_a_lds(st, LDSS, arow, 0, lane);
    v16bf sa1 = load_a_lds(st, LDSS, arow, 32, lane);
    v16bf ha0 = load_a_lds(sh_h, LDH, arow, 0, lane);
    v16bf ha1 = load_a_lds(sh_h, LDH, arow, 32, lane);

    v8f au, arc, ag, ahg;
#pragma unroll
    for (int r = 0; r < 8; ++r) { au[r] = bu; arc[r] = brr; ag[r] = bg; ahg[r] = 0.0f; }
    au  = wmma_bf16(sa0, xu0, au);  au  = wmma_bf16(sa1, xu1, au);
    au  = wmma_bf16(ha0, hu0, au);  au  = wmma_bf16(ha1, hu1, au);
    arc = wmma_bf16(sa0, xr0, arc); arc = wmma_bf16(sa1, xr1, arc);
    arc = wmma_bf16(ha0, hr0, arc); arc = wmma_bf16(ha1, hr1, arc);
    ag  = wmma_bf16(sa0, xg0, ag);  ag  = wmma_bf16(sa1, xg1, ag);
    ahg = wmma_bf16(ha0, hg0, ahg); ahg = wmma_bf16(ha1, hg1, ahg);

    float hn[8];
#pragma unroll
    for (int r = 0; r < 8; ++r) {
      int M = mrow + r;
      float hold = sh_hf[M * H_SZ + cl];
      float a_t = sh_sc[t * ROWS + M];
      float u = sigmoidf_(au[r]) * a_t;
      float rr = sigmoidf_(arc[r]);
      float g = tanhf(ag[r] + rr * ahg[r]);
      float v = (1.0f - u) * hold + u * g;
      unsigned char m = mask[(size_t)(b0 + M) * T_SZ + t];
      hn[r] = m ? v : hold;
    }
    __syncthreads();  // all cross-slice h reads complete before overwriting h
#pragma unroll
    for (int r = 0; r < 8; ++r) {
      int M = mrow + r;
      sh_hf[M * H_SZ + cl] = hn[r];
      sh_h[M * LDH + cl] = (__bf16)hn[r];
      out[((size_t)(b0 + M) * T_SZ + t) * H_SZ + cl] = hn[r];
    }
  }
}

extern "C" void kernel_launch(void* const* d_in, const int* in_sizes, int n_in,
                              void* d_out, int out_size, void* d_ws, size_t ws_size,
                              hipStream_t stream) {
  (void)in_sizes; (void)n_in; (void)out_size; (void)ws_size;
  const float* behaviors    = (const float*)d_in[0];
  const float* candidate    = (const float*)d_in[1];
  const unsigned char* mask = (const unsigned char*)d_in[2];  // numpy bool = 1B
  const float* gru_kernel   = (const float*)d_in[3];
  const float* gru_rec      = (const float*)d_in[4];
  const float* gru_bias     = (const float*)d_in[5];
  const float* W_kernel     = (const float*)d_in[6];
  const float* W_bias       = (const float*)d_in[7];
  const float* Wxu          = (const float*)d_in[8];
  const float* bxu          = (const float*)d_in[9];
  const float* Whu          = (const float*)d_in[10];
  const float* Wxr          = (const float*)d_in[11];
  const float* bxr          = (const float*)d_in[12];
  const float* Whr          = (const float*)d_in[13];
  const float* Wxg          = (const float*)d_in[14];
  const float* bxg          = (const float*)d_in[15];
  const float* Whg          = (const float*)d_in[16];

  dien_fused<<<dim3(B_SZ / ROWS), dim3(NTHR), 0, stream>>>(
      behaviors, candidate, mask, gru_kernel, gru_rec, gru_bias, W_kernel, W_bias,
      Wxu, bxu, Whu, Wxr, bxr, Whr, Wxg, bxg, Whg, (float*)d_out, d_ws);
}
